// SelfAttention_84885733638238
// MI455X (gfx1250) — matrix operand order repaired and checked by reading, not compiled, not run on hardware
//
#include <hip/hip_runtime.h>
#include <math.h>

#define DEV __device__ __forceinline__

typedef __bf16 bf16;
typedef __attribute__((ext_vector_type(16))) __bf16 bf16x16;
typedef __attribute__((ext_vector_type(8)))  float  floatx8;

constexpr int Bn = 4, S = 1024, D = 1024, H = 16, HD = 64;
constexpr int BS = Bn * S;          // 4096 rows

// ---------- small helpers ----------
DEV bf16 f2bf(float f) {            // round-to-nearest-even fp32 -> bf16
  union { float f; unsigned u; } in; in.f = f;
  unsigned r = (in.u + 0x7FFFu + ((in.u >> 16) & 1u)) >> 16;
  union { unsigned short s; bf16 b; } out; out.s = (unsigned short)r;
  return out.b;
}
DEV unsigned short bfbits(bf16 b) {
  union { bf16 b; unsigned short s; } u; u.b = b; return u.s;
}

// A-fragment: 16(M) x 32(K) bf16, source row-major (K contiguous), stride ld.
// ISA: lanes 0-15 rows M; VGPR0-3 -> K0..7 (half0)/8..15 (half1); VGPR4-7 -> +16.
DEV bf16x16 load_fragA_rm(const bf16* base, int ld, int lane) {
  int m = lane & 15, half = lane >> 4;
  union { unsigned u[8]; bf16x16 v; } f;
  const bf16* row = base + (size_t)m * ld;
#pragma unroll
  for (int v = 0; v < 8; ++v) {
    int k = 2 * (v & 3) + ((v >> 2) << 4) + (half << 3);
    f.u[v] = *reinterpret_cast<const unsigned*>(row + k);   // pair K,K+1
  }
  return f.v;
}

// B-fragment: 32(K) x 16(N), source row-major src[N][K] (B = src^T).
// lanes hold column N; VGPR0-3 -> K0..7 (half0)/8..15 (half1); VGPR4-7 -> +16.
DEV bf16x16 load_fragB_nk(const bf16* base, int ld, int lane) {
  int n = lane & 15, half = lane >> 4;
  union { unsigned u[8]; bf16x16 v; } f;
  const bf16* row = base + (size_t)n * ld;
#pragma unroll
  for (int v = 0; v < 8; ++v) {
    int k = 2 * (v & 3) + ((v >> 2) << 4) + (half << 3);
    f.u[v] = *reinterpret_cast<const unsigned*>(row + k);
  }
  return f.v;
}

DEV floatx8 wmma_bf16(bf16x16 a, bf16x16 b, floatx8 c) {
  return __builtin_amdgcn_wmma_f32_16x16x32_bf16(false, a, false, b,
                                                 (short)0, c, false, false);
}

// ---------- kernel 1: fp32 -> bf16 (8 elems/thread, b128 in/out) ----------
__global__ __launch_bounds__(256) void cvt_bf16(const float* __restrict__ in,
                                                bf16* __restrict__ out, int n) {
  int i = (blockIdx.x * 256 + threadIdx.x) * 8;
  if (i >= n) return;
  float4 f0 = *reinterpret_cast<const float4*>(in + i);
  float4 f1 = *reinterpret_cast<const float4*>(in + i + 4);
  union { unsigned short us[8]; uint4 q; } pk;
  pk.us[0] = bfbits(f2bf(f0.x)); pk.us[1] = bfbits(f2bf(f0.y));
  pk.us[2] = bfbits(f2bf(f0.z)); pk.us[3] = bfbits(f2bf(f0.w));
  pk.us[4] = bfbits(f2bf(f1.x)); pk.us[5] = bfbits(f2bf(f1.y));
  pk.us[6] = bfbits(f2bf(f1.z)); pk.us[7] = bfbits(f2bf(f1.w));
  *reinterpret_cast<uint4*>(out + i) = pk.q;
}

// ---------- shared GEMM plumbing ------------------------------------------
DEV void load_gemm_frags(const bf16* Abase, const bf16* W, int nbase, int kk,
                         int lane, bf16x16& a0, bf16x16& a1, bf16x16 bx[4]) {
  a0 = load_fragA_rm(Abase + kk, D, lane);
  a1 = load_fragA_rm(Abase + (size_t)16 * D + kk, D, lane);
#pragma unroll
  for (int t = 0; t < 4; ++t)
    bx[t] = load_fragB_nk(W + (size_t)(nbase + 16 * t) * D + kk, D, lane);
}

DEV void gemm_step(const bf16x16& a0, const bf16x16& a1, const bf16x16 b[4],
                   floatx8 acc[2][4]) {
#pragma unroll
  for (int t = 0; t < 4; ++t) acc[0][t] = wmma_bf16(a0, b[t], acc[0][t]);
#pragma unroll
  for (int t = 0; t < 4; ++t) acc[1][t] = wmma_bf16(a1, b[t], acc[1][t]);
}

// ---------- kernel 2: QKV projection -------------------------------------
// grid (BS/32, D/512, 3), block 256 (8 waves); wave -> 32(M) x 64(N) tile.
// K-loop unrolled x2 with two register buffer sets (no rotation copies).
// z==0/1 (Q,K): head layout [b,h,s,hd] bf16. z==2 (V): TRANSPOSED [b,h,hd,s].
__global__ __launch_bounds__(256) void qkv_gemm(
    const bf16* __restrict__ xb,
    const bf16* __restrict__ wq, const bf16* __restrict__ wk,
    const bf16* __restrict__ wv,
    const float* __restrict__ bq, const float* __restrict__ bk,
    const float* __restrict__ bv, bf16* __restrict__ qkv) {
  int wave = threadIdx.x >> 5, lane = threadIdx.x & 31;
  int z = blockIdx.z;
  const bf16*  W    = (z == 0) ? wq : (z == 1) ? wk : wv;
  const float* bias = (z == 0) ? bq : (z == 1) ? bk : bv;
  bf16* out = qkv + (size_t)z * BS * D;

  int mbase = blockIdx.x * 32;
  int nbase = blockIdx.y * 512 + wave * 64;

  floatx8 z8 = (floatx8)0.0f;
  floatx8 acc[2][4] = {{z8, z8, z8, z8}, {z8, z8, z8, z8}};

  const bf16* Abase = xb + (size_t)mbase * D;
  bf16x16 a0[2], a1[2], b[2][4];
  load_gemm_frags(Abase, W, nbase, 0, lane, a0[0], a1[0], b[0]);

  for (int kk = 0; kk < D; kk += 64) {
    load_gemm_frags(Abase, W, nbase, kk + 32, lane, a0[1], a1[1], b[1]);
    __builtin_prefetch(W + (size_t)nbase * D + kk + 512, 0, 1);
    gemm_step(a0[0], a1[0], b[0], acc);
    int k2 = (kk + 64 < D) ? kk + 64 : 0;   // last prefetch wraps (discarded)
    load_gemm_frags(Abase, W, nbase, k2, lane, a0[0], a1[0], b[0]);
    gemm_step(a0[1], a1[1], b[1], acc);
  }

  int n0 = lane & 15, half = lane >> 4;
#pragma unroll
  for (int t = 0; t < 4; ++t) {
    int n = nbase + 16 * t + n0;
    int h = n >> 6, hd = n & 63;
    float bias_n = bias[n];
#pragma unroll
    for (int g = 0; g < 2; ++g) {
      if (z == 2) {
        // V transposed: [b, h, hd, s]; 8 consecutive s per lane -> b128 store
        int row0 = mbase + 16 * g + 8 * half;
        int bb = row0 >> 10, s0 = row0 & (S - 1);
        union { unsigned short us[8]; uint4 q; } pk;
#pragma unroll
        for (int r = 0; r < 8; ++r)
          pk.us[r] = bfbits(f2bf(acc[g][t][r] + bias_n));
        *reinterpret_cast<uint4*>(
            out + (((size_t)bb * H + h) * HD + hd) * S + s0) = pk.q;
      } else {
#pragma unroll
        for (int r = 0; r < 8; ++r) {
          int row = mbase + 16 * g + r + 8 * half;
          int bb = row >> 10, s = row & (S - 1);
          out[(((size_t)bb * H + h) * S + s) * HD + hd] =
              f2bf(acc[g][t][r] + bias_n);
        }
      }
    }
  }
}

// ---------- kernel 3: flash attention (softmax then /8), + residual --------
// grid = B*H*(S/64), block 128 (4 waves); each wave owns 16 query rows.
// Key loop unrolled x2 with two K-fragment buffer sets.
__global__ __launch_bounds__(128) void attention(
    const bf16* __restrict__ qkv, const float* __restrict__ x,
    bf16* __restrict__ attx) {
  __shared__ bf16 Pl[4][16][34];               // per-wave P tile, padded rows
  int wave = threadIdx.x >> 5, lane = threadIdx.x & 31;
  int bh = blockIdx.x >> 4;                    // S/64 = 16 q-blocks per head
  int qt = ((blockIdx.x & 15) << 2) + wave;    // 16-row tile id, 0..63
  int b = bh >> 4, h = bh & 15;
  const size_t hsz = (size_t)BS * D;
  const bf16* Q  = qkv + (size_t)bh * S * HD;
  const bf16* K  = qkv + hsz + (size_t)bh * S * HD;
  const bf16* Vt = qkv + 2 * hsz + (size_t)bh * HD * S;   // transposed [hd, s]
  int q0 = qt * 16;

  bf16x16 aq0 = load_fragA_rm(Q + (size_t)q0 * HD,      HD, lane);
  bf16x16 aq1 = load_fragA_rm(Q + (size_t)q0 * HD + 32, HD, lane);

  floatx8 z8 = (floatx8)0.0f;
  floatx8 o[4] = {z8, z8, z8, z8};
  float mrow[8], lrow[8];
#pragma unroll
  for (int r = 0; r < 8; ++r) { mrow[r] = -INFINITY; lrow[r] = 0.0f; }

  int half = lane >> 4, n0 = lane & 15;

  auto load_k_frags = [&](int kt, bf16x16 bk[4]) {
    bk[0] = load_fragB_nk(K + (size_t)kt * HD, HD, lane);
    bk[1] = load_fragB_nk(K + (size_t)kt * HD + 32, HD, lane);
    bk[2] = load_fragB_nk(K + (size_t)(kt + 16) * HD, HD, lane);
    bk[3] = load_fragB_nk(K + (size_t)(kt + 16) * HD + 32, HD, lane);
  };

  auto step = [&](int kt, const bf16x16 bk[4]) {
    // scores for two 16-key subtiles (K-dim = HD = 2 x 32)
    floatx8 sc[2];
    floatx8 s0 = wmma_bf16(aq0, bk[0], z8);
    sc[0] = wmma_bf16(aq1, bk[1], s0);
    floatx8 s1 = wmma_bf16(aq0, bk[2], z8);
    sc[1] = wmma_bf16(aq1, bk[3], s1);

    // V fragments for this tile (independent of P) -- issue early
    bf16x16 bv[4];
#pragma unroll
    for (int t = 0; t < 4; ++t)
      bv[t] = load_fragB_nk(Vt + (size_t)(16 * t) * S + kt, S, lane);

    // online softmax; accumulator row m = r + 8*half lives on this lane group
#pragma unroll
    for (int r = 0; r < 8; ++r) {
      float vmax = fmaxf(sc[0][r], sc[1][r]);
#pragma unroll
      for (int msk = 8; msk >= 1; msk >>= 1)
        vmax = fmaxf(vmax, __shfl_xor(vmax, msk, 16));
      float mnew  = fmaxf(mrow[r], vmax);
      float scale = __expf(mrow[r] - mnew);
      float p0 = __expf(sc[0][r] - mnew);
      float p1 = __expf(sc[1][r] - mnew);
      float ps = p0 + p1;
#pragma unroll
      for (int msk = 8; msk >= 1; msk >>= 1) ps += __shfl_xor(ps, msk, 16);
      lrow[r] = lrow[r] * scale + ps;
      mrow[r] = mnew;
      o[0][r] *= scale; o[1][r] *= scale; o[2][r] *= scale; o[3][r] *= scale;
      Pl[wave][r + 8 * half][n0]      = f2bf(p0);
      Pl[wave][r + 8 * half][16 + n0] = f2bf(p1);
    }
    // P (16x32) back as A-fragment from LDS, then O += P * V
    bf16x16 ap = load_fragA_rm(&Pl[wave][0][0], 34, lane);
#pragma unroll
    for (int t = 0; t < 4; ++t) o[t] = wmma_bf16(ap, bv[t], o[t]);
  };

  bf16x16 bkbuf[2][4];
  load_k_frags(0, bkbuf[0]);
  for (int kt = 0; kt < S; kt += 64) {
    load_k_frags(kt + 32, bkbuf[1]);
    step(kt, bkbuf[0]);
    int kn = (kt + 64 < S) ? kt + 64 : 0;   // last prefetch wraps (discarded)
    load_k_frags(kn, bkbuf[0]);
    step(kt + 32, bkbuf[1]);
  }

  // normalize (softmax sum * sqrt(HD)=8), add residual x, store bf16
#pragma unroll
  for (int t = 0; t < 4; ++t) {
#pragma unroll
    for (int r = 0; r < 8; ++r) {
      int srow = q0 + r + 8 * half;
      int col  = h * HD + 16 * t + n0;
      size_t idx = ((size_t)(b * S + srow)) * D + col;
      float val = o[t][r] / (lrow[r] * 8.0f) + x[idx];
      attx[idx] = f2bf(val);
    }
  }
}

// ---------- kernel 4: output linear y = attx @ wl^T + bl (fp32 out) --------
// Same pipelined 32x64 wave tile as qkv_gemm. grid (BS/32, D/512).
__global__ __launch_bounds__(256) void out_gemm(
    const bf16* __restrict__ attx, const bf16* __restrict__ wl,
    const float* __restrict__ bl, float* __restrict__ y) {
  int wave = threadIdx.x >> 5, lane = threadIdx.x & 31;
  int mbase = blockIdx.x * 32;
  int nbase = blockIdx.y * 512 + wave * 64;

  floatx8 z8 = (floatx8)0.0f;
  floatx8 acc[2][4] = {{z8, z8, z8, z8}, {z8, z8, z8, z8}};

  const bf16* Abase = attx + (size_t)mbase * D;
  bf16x16 a0[2], a1[2], b[2][4];
  load_gemm_frags(Abase, wl, nbase, 0, lane, a0[0], a1[0], b[0]);

  for (int kk = 0; kk < D; kk += 64) {
    load_gemm_frags(Abase, wl, nbase, kk + 32, lane, a0[1], a1[1], b[1]);
    __builtin_prefetch(wl + (size_t)nbase * D + kk + 512, 0, 1);
    gemm_step(a0[0], a1[0], b[0], acc);
    int k2 = (kk + 64 < D) ? kk + 64 : 0;
    load_gemm_frags(Abase, wl, nbase, k2, lane, a0[0], a1[0], b[0]);
    gemm_step(a0[1], a1[1], b[1], acc);
  }

  int n0 = lane & 15, half = lane >> 4;
#pragma unroll
  for (int t = 0; t < 4; ++t) {
    int n = nbase + 16 * t + n0;
    float bias_n = bl[n];
#pragma unroll
    for (int g = 0; g < 2; ++g)
#pragma unroll
      for (int r = 0; r < 8; ++r) {
        int row = mbase + 16 * g + r + 8 * half;
        y[(size_t)row * D + n] = acc[g][t][r] + bias_n;
      }
  }
}

// ---------- kernel 5: LayerNorm over rows of 1024 --------------------------
__global__ __launch_bounds__(256) void layernorm(
    const float* __restrict__ y, const float* __restrict__ g,
    const float* __restrict__ be, float* __restrict__ out) {
  __shared__ float red[2][8];
  int row = blockIdx.x;
  const float* yr = y + (size_t)row * D;
  int i4 = threadIdx.x * 4;
  float4 v = *reinterpret_cast<const float4*>(yr + i4);
  float s  = v.x + v.y + v.z + v.w;
  float s2 = v.x * v.x + v.y * v.y + v.z * v.z + v.w * v.w;
#pragma unroll
  for (int msk = 16; msk >= 1; msk >>= 1) {
    s  += __shfl_xor(s,  msk, 32);
    s2 += __shfl_xor(s2, msk, 32);
  }
  int wave = threadIdx.x >> 5, lane = threadIdx.x & 31;
  if (lane == 0) { red[0][wave] = s; red[1][wave] = s2; }
  __syncthreads();
  float ts = 0.f, ts2 = 0.f;
#pragma unroll
  for (int w = 0; w < 8; ++w) { ts += red[0][w]; ts2 += red[1][w]; }
  float mu  = ts / (float)D;
  float var = ts2 / (float)D - mu * mu;
  float inv = rsqrtf(var + 1e-5f);
  float4 gg = *reinterpret_cast<const float4*>(g + i4);
  float4 bb = *reinterpret_cast<const float4*>(be + i4);
  float4 o;
  o.x = (v.x - mu) * inv * gg.x + bb.x;
  o.y = (v.y - mu) * inv * gg.y + bb.y;
  o.z = (v.z - mu) * inv * gg.z + bb.z;
  o.w = (v.w - mu) * inv * gg.w + bb.w;
  *reinterpret_cast<float4*>(out + (size_t)row * D + i4) = o;
}

// ---------- host launcher ---------------------------------------------------
extern "C" void kernel_launch(void* const* d_in, const int* in_sizes, int n_in,
                              void* d_out, int out_size, void* d_ws,
                              size_t ws_size, hipStream_t stream) {
  const float* x    = (const float*)d_in[0];
  const float* wq   = (const float*)d_in[1];
  const float* bq   = (const float*)d_in[2];
  const float* wk   = (const float*)d_in[3];
  const float* bk   = (const float*)d_in[4];
  const float* wv   = (const float*)d_in[5];
  const float* bv   = (const float*)d_in[6];
  const float* wl   = (const float*)d_in[7];
  const float* bl   = (const float*)d_in[8];
  const float* ln_g = (const float*)d_in[9];
  const float* ln_b = (const float*)d_in[10];

  char* ws = (char*)d_ws;
  const size_t MiB = 1024 * 1024;
  bf16*  xb   = (bf16*)(ws + 0);          //  8 MiB : x bf16 [BS, D]
  bf16*  wqb  = (bf16*)(ws + 8  * MiB);   //  2 MiB
  bf16*  wkb  = (bf16*)(ws + 10 * MiB);   //  2 MiB
  bf16*  wvb  = (bf16*)(ws + 12 * MiB);   //  2 MiB
  bf16*  wlb  = (bf16*)(ws + 14 * MiB);   //  2 MiB
  bf16*  qkvb = (bf16*)(ws + 16 * MiB);   // 24 MiB : Q,K [b,h,s,hd]; V [b,h,hd,s]
  bf16*  attx = (bf16*)(ws + 40 * MiB);   //  8 MiB : (att + x) bf16
  float* yws  = (float*)(ws + 48 * MiB);  // 16 MiB : pre-LN fp32
  (void)in_sizes; (void)n_in; (void)out_size; (void)ws_size;

  cvt_bf16<<<(BS * D) / 2048, 256, 0, stream>>>(x,  xb,  BS * D);
  cvt_bf16<<<(D * D) / 2048,  256, 0, stream>>>(wq, wqb, D * D);
  cvt_bf16<<<(D * D) / 2048,  256, 0, stream>>>(wk, wkb, D * D);
  cvt_bf16<<<(D * D) / 2048,  256, 0, stream>>>(wv, wvb, D * D);
  cvt_bf16<<<(D * D) / 2048,  256, 0, stream>>>(wl, wlb, D * D);

  dim3 g1(BS / 32, D / 512, 3);
  qkv_gemm<<<g1, 256, 0, stream>>>(xb, wqb, wkb, wvb, bq, bk, bv, qkvb);

  attention<<<Bn * H * (S / 64), 128, 0, stream>>>(qkvb, x, attx);

  dim3 g2(BS / 32, D / 512);
  out_gemm<<<g2, 256, 0, stream>>>(attx, wlb, bl, yws);

  layernorm<<<BS, 256, 0, stream>>>(yws, ln_g, ln_b, (float*)d_out);
}
